// Pointnet2Backbone_MSG_7937099563184
// MI455X (gfx1250) — matrix operand order repaired and checked by reading, not compiled, not run on hardware
//
#include <hip/hip_runtime.h>
#include <hip/hip_bf16.h>
#include <cstdint>
#include <cstddef>

// ---------------------------------------------------------------------------
// CDNA5 (gfx1250) wave32 WMMA types
// ---------------------------------------------------------------------------
typedef __attribute__((ext_vector_type(16))) __bf16 v16bf;
typedef __attribute__((ext_vector_type(8)))  __bf16 v8bf;
typedef __attribute__((ext_vector_type(8)))  float  v8f;
typedef __attribute__((ext_vector_type(4)))  int    athena_v4i;

#if __has_builtin(__builtin_amdgcn_global_load_async_to_lds_b128)
#define ATHENA_ASYNC_LDS 1
// AS-qualified pointer types matching the builtin's prototype
typedef __attribute__((address_space(1))) athena_v4i* athena_as1_v4i;
typedef __attribute__((address_space(3))) athena_v4i* athena_as3_v4i;
typedef __attribute__((address_space(1))) void*       athena_as1_void;
typedef __attribute__((address_space(3))) void*       athena_as3_void;

static __device__ __forceinline__ void athena_wait_async0() {
#if __has_builtin(__builtin_amdgcn_s_wait_asynccnt)
  __builtin_amdgcn_s_wait_asynccnt(0);
#else
  asm volatile("s_wait_asynccnt 0x0" ::: "memory");
#endif
}
#endif

// ---------------------------------------------------------------------------
// Farthest point sampling: one block per batch, serial over npoint iterations,
// parallel min-dist update + argmax reduction (ties -> lowest index).
// ---------------------------------------------------------------------------
__global__ void fps_kernel(const float* __restrict__ xyz, int N, int npoint,
                           int* __restrict__ inds, float* __restrict__ dist) {
  int b = blockIdx.x;
  xyz  += (size_t)b * N * 3;
  inds += (size_t)b * npoint;
  dist += (size_t)b * N;
  int tid = threadIdx.x, nt = blockDim.x;
  for (int i = tid; i < N; i += nt) dist[i] = 1e10f;
  __shared__ float sv[256];
  __shared__ int   si[256];
  __shared__ int   s_last;
  if (tid == 0) s_last = 0;
  __syncthreads();
  for (int j = 0; j < npoint; ++j) {
    int last = s_last;
    if (tid == 0) inds[j] = last;
    float lx = xyz[last * 3 + 0];
    float ly = xyz[last * 3 + 1];
    float lz = xyz[last * 3 + 2];
    float bestv = -1.0f;
    int   besti = 0;
    for (int i = tid; i < N; i += nt) {
      float dx = xyz[i * 3 + 0] - lx;
      float dy = xyz[i * 3 + 1] - ly;
      float dz = xyz[i * 3 + 2] - lz;
      float d  = dx * dx + dy * dy + dz * dz;
      float nd = fminf(dist[i], d);
      dist[i] = nd;
      if (nd > bestv) { bestv = nd; besti = i; }
    }
    sv[tid] = bestv; si[tid] = besti;
    __syncthreads();
    for (int s = nt >> 1; s > 0; s >>= 1) {
      if (tid < s) {
        if (sv[tid + s] > sv[tid] ||
            (sv[tid + s] == sv[tid] && si[tid + s] < si[tid])) {
          sv[tid] = sv[tid + s]; si[tid] = si[tid + s];
        }
      }
      __syncthreads();
    }
    if (tid == 0) s_last = si[0];
    __syncthreads();
  }
}

// Gather sampled centers: out[b,p,:] = xyz[b, inds[b,p], :]
__global__ void gather_xyz_kernel(const float* __restrict__ xyz,
                                  const int* __restrict__ inds,
                                  float* __restrict__ out,
                                  int npoint, int N, int total) {
  int t = blockIdx.x * blockDim.x + threadIdx.x;
  if (t >= total) return;
  int b = t / npoint;
  int id = inds[t];
  const float* s = xyz + ((size_t)b * N + id) * 3;
  float* d = out + (size_t)t * 3;
  d[0] = s[0]; d[1] = s[1]; d[2] = s[2];
}

// Ball query matching reference semantics: first `nsample` source indices
// (increasing) with d2 <= r2, padded with the first hit.
__global__ void ball_query_kernel(const float* __restrict__ xyz,
                                  const float* __restrict__ qxyz,
                                  int N, int P, float r2, int nsample,
                                  int* __restrict__ idx) {
  int p = blockIdx.x * blockDim.x + threadIdx.x;
  if (p >= P) return;
  float qx = qxyz[p * 3 + 0], qy = qxyz[p * 3 + 1], qz = qxyz[p * 3 + 2];
  int* row = idx + (size_t)p * nsample;
  int cnt = 0, first = N - 1;
  for (int i = 0; i < N && cnt < nsample; ++i) {
    float dx = xyz[i * 3 + 0] - qx;
    float dy = xyz[i * 3 + 1] - qy;
    float dz = xyz[i * 3 + 2] - qz;
    if (dx * dx + dy * dy + dz * dz <= r2) {
      if (cnt == 0) first = i;
      row[cnt++] = i;
    }
  }
  for (; cnt < nsample; ++cnt) row[cnt] = first;
}

// Grouping with padded row stride Ctp (pad channels zero-filled):
// g[p,s,0:3] = xyz[idx]-center ; g[p,s,3:3+C] = feats[idx] ; g[p,s,3+C:Ctp]=0
__global__ void group_kernel(const float* __restrict__ xyz,
                             const float* __restrict__ feats,
                             const int* __restrict__ idx,
                             const float* __restrict__ qxyz,
                             float* __restrict__ g,
                             int P, int nsample, int C, int Ctp) {
  int t = blockIdx.x * blockDim.x + threadIdx.x;
  if (t >= P * nsample) return;
  int p  = t / nsample;
  int id = idx[t];
  int Ct = 3 + C;
  float* o = g + (size_t)t * Ctp;
  o[0] = xyz[id * 3 + 0] - qxyz[p * 3 + 0];
  o[1] = xyz[id * 3 + 1] - qxyz[p * 3 + 1];
  o[2] = xyz[id * 3 + 2] - qxyz[p * 3 + 2];
  for (int c = 0; c < C; ++c) o[3 + c] = feats[(size_t)id * C + c];
  for (int c = Ct; c < Ctp; ++c) o[c] = 0.0f;
}

// ---------------------------------------------------------------------------
// Weight pre-swizzle: fp32 (K,N) row-major -> bf16 in this file's contiguous-k
// B-fragment order. Element t: e = t&15, lane = (t>>4)&31, blk = t>>9 with
// blk = kb*nTiles + tn;  k = kb*32 + (lane>>4)*16 + e ; n = tn*16 + (lane&15).
// Zero-fills the (Kp,Np) padding so the GEMM loop needs no guards.
// ---------------------------------------------------------------------------
__global__ void pack_w_kernel(const float* __restrict__ W, __bf16* __restrict__ Wp,
                              int K, int N, int Kp, int Np, int total) {
  int t = blockIdx.x * blockDim.x + threadIdx.x;
  if (t >= total) return;
  int e    = t & 15;
  int lane = (t >> 4) & 31;
  int blk  = t >> 9;
  int nTiles = Np >> 4;
  int kb = blk / nTiles, tn = blk - kb * nTiles;
  int k = kb * 32 + (lane >> 4) * 16 + e;
  int n = tn * 16 + (lane & 15);
  float v = (k < K && n < N) ? W[(size_t)k * N + n] : 0.0f;
  Wp[t] = (__bf16)v;
}

// ---------------------------------------------------------------------------
// GEMM + bias + ReLU via V_WMMA_F32_16X16X32_BF16, wave32, no guards.
// Block = 4 waves; each wave owns a 16-row strip; the block shares one
// 64-column weight panel, staged K-chunk by K-chunk into LDS (32 KB) with
// GLOBAL_LOAD_ASYNC_TO_LDS_B128 when available (ASYNCcnt path), else a
// register copy. Per K-step per wave:
//   A: 4x global_load_b128 (fp32, padded stride, pad zeroed) -> cvt_pk_bf16
//   B: 2x ds_load_b128 per tile from the swizzled LDS panel
//   4 accumulators (16x64 strip) share one A fragment.
// ---------------------------------------------------------------------------
#define ATHENA_KC 256   // K-chunk staged per LDS round

__global__ void __launch_bounds__(128)
gemm_bias_relu_wmma(const float* __restrict__ A, const __bf16* __restrict__ Wp,
                    const float* __restrict__ bias, float* __restrict__ Cmat,
                    int Kp, int N, int Np) {
  __shared__ __bf16 sB[ATHENA_KC / 32][4][32][16];   // 32 KB swizzled panel
  const int tid  = threadIdx.x;
  const int wave = tid >> 5;
  const int lane = tid & 31;
  const int tm   = blockIdx.x * 4 + wave;
  const int tg   = blockIdx.y;                        // group of 4 n-tiles
  const int nTiles = Np >> 4;
  const int half = lane >> 4;
  const int q    = lane & 15;
  const int row  = tm * 16 + q;
  const float* __restrict__ arow = A + (size_t)row * Kp;

  v8f acc[4];
#pragma unroll
  for (int t = 0; t < 4; ++t) acc[t] = (v8f){};

  for (int k0 = 0; k0 < Kp; k0 += ATHENA_KC) {
    const int krem   = Kp - k0;
    const int ksteps = (krem < ATHENA_KC ? krem : ATHENA_KC) >> 5;

    // ---- stage the (ksteps*32) x 64 weight panel into LDS ----
    const int units = ksteps * 256;                   // 16-byte units
    for (int u = tid; u < units; u += 128) {
      int uh    = u & 1;
      int lane2 = (u >> 1) & 31;
      int t     = (u >> 6) & 3;
      int kbl   = u >> 8;
      int lofs  = ((kbl * 4 + t) * 32 + lane2) * 16 + uh * 8;
      size_t gofs = ((size_t)((k0 >> 5) + kbl) * nTiles + (tg * 4 + t)) * 512
                    + lane2 * 16 + uh * 8;
      const __bf16* src = Wp + gofs;
      __bf16* dst = &sB[0][0][0][0] + lofs;
#if defined(ATHENA_ASYNC_LDS)
      __builtin_amdgcn_global_load_async_to_lds_b128(
          (athena_as1_v4i)(athena_as1_void)(void*)src,
          (athena_as3_v4i)(athena_as3_void)(void*)dst, 0, 0);
#else
      *(v8bf*)dst = *(const v8bf*)src;
#endif
    }
#if defined(ATHENA_ASYNC_LDS)
    athena_wait_async0();
#endif
    __syncthreads();

    // ---- consume the chunk ----
    for (int kbl = 0; kbl < ksteps; ++kbl) {
      const int kk = k0 + (kbl << 5);
      __builtin_prefetch(arow + kk + ATHENA_KC, 0, 1);   // -> global_prefetch_b8
      const float4* ap0 = (const float4*)(arow + kk + half * 16);
      const float4* ap1 = (const float4*)(arow + kk + half * 16 + 8);
      float4 a0 = ap0[0], a1 = ap0[1];
      float4 a2 = ap1[0], a3 = ap1[1];
      v16bf afrag;
      afrag[0]  = (__bf16)a0.x; afrag[1]  = (__bf16)a0.y;
      afrag[2]  = (__bf16)a0.z; afrag[3]  = (__bf16)a0.w;
      afrag[4]  = (__bf16)a1.x; afrag[5]  = (__bf16)a1.y;
      afrag[6]  = (__bf16)a1.z; afrag[7]  = (__bf16)a1.w;
      afrag[8]  = (__bf16)a2.x; afrag[9]  = (__bf16)a2.y;
      afrag[10] = (__bf16)a2.z; afrag[11] = (__bf16)a2.w;
      afrag[12] = (__bf16)a3.x; afrag[13] = (__bf16)a3.y;
      afrag[14] = (__bf16)a3.z; afrag[15] = (__bf16)a3.w;
#pragma unroll
      for (int t = 0; t < 4; ++t) {
        v8bf b0 = *(const v8bf*)&sB[kbl][t][lane][0];
        v8bf b1 = *(const v8bf*)&sB[kbl][t][lane][8];
        v16bf bfrag = __builtin_shufflevector(b0, b1,
            0, 1, 2, 3, 4, 5, 6, 7, 8, 9, 10, 11, 12, 13, 14, 15);
        acc[t] = __builtin_amdgcn_wmma_f32_16x16x32_bf16(
            /*neg_a=*/false, afrag, /*neg_b=*/false, bfrag,
            /*c_mod=*/(short)0, acc[t], /*reuse_a=*/false, /*reuse_b=*/false);
      }
    }
    __syncthreads();
  }

  const int mrow = tm * 16 + half * 8;
#pragma unroll
  for (int t = 0; t < 4; ++t) {
    int col = (tg * 4 + t) * 16 + q;
    float bv = (col < N) ? bias[col] : 0.0f;
    float* out = Cmat + (size_t)mrow * Np + col;
#pragma unroll
    for (int r = 0; r < 8; ++r) {
      float v = acc[t][r] + bv;
      v = v > 0.0f ? v : 0.0f;
      if (col >= N) v = 0.0f;               // keep pad columns zero
      out[(size_t)r * Np] = v;
    }
  }
}

// Max over the nsample axis (input rows have padded stride inStride):
// out[p, c] = max_s h[(p*ns+s)*inStride + c]
__global__ void maxpool_kernel(const float* __restrict__ h,
                               float* __restrict__ out,
                               int P, int nsample, int C,
                               int inStride, int outStride) {
  int t = blockIdx.x * blockDim.x + threadIdx.x;
  if (t >= P * C) return;
  int p = t / C, c = t % C;
  float m = -1e30f;
  for (int s = 0; s < nsample; ++s)
    m = fmaxf(m, h[((size_t)p * nsample + s) * inStride + c]);
  out[(size_t)p * outStride + c] = m;
}

// Feature propagation: 3-NN inverse-squared-distance interpolation and concat
__global__ void fp_interp_concat_kernel(const float* __restrict__ ux,
                                        const float* __restrict__ kx,
                                        const float* __restrict__ uf,
                                        const float* __restrict__ kf,
                                        float* __restrict__ h,
                                        int U, int Kn, int Cu, int Ck) {
  int u = blockIdx.x * blockDim.x + threadIdx.x;
  if (u >= U) return;
  float qx = ux[u * 3 + 0], qy = ux[u * 3 + 1], qz = ux[u * 3 + 2];
  float d0 = 1e30f, d1 = 1e30f, d2 = 1e30f;
  int   i0 = 0, i1 = 0, i2 = 0;
  for (int i = 0; i < Kn; ++i) {
    float dx = kx[i * 3 + 0] - qx;
    float dy = kx[i * 3 + 1] - qy;
    float dz = kx[i * 3 + 2] - qz;
    float d = dx * dx + dy * dy + dz * dz;
    if (d < d0)      { d2 = d1; i2 = i1; d1 = d0; i1 = i0; d0 = d; i0 = i; }
    else if (d < d1) { d2 = d1; i2 = i1; d1 = d;  i1 = i; }
    else if (d < d2) { d2 = d;  i2 = i; }
  }
  float w0 = 1.0f / (d0 + 1e-8f);
  float w1 = 1.0f / (d1 + 1e-8f);
  float w2 = 1.0f / (d2 + 1e-8f);
  float ws = w0 + w1 + w2;
  w0 /= ws; w1 /= ws; w2 /= ws;
  float* hr = h + (size_t)u * (Cu + Ck);
  for (int c = 0; c < Cu; ++c) hr[c] = uf[(size_t)u * Cu + c];
  for (int c = 0; c < Ck; ++c)
    hr[Cu + c] = w0 * kf[(size_t)i0 * Ck + c] +
                 w1 * kf[(size_t)i1 * Ck + c] +
                 w2 * kf[(size_t)i2 * Ck + c];
}

__global__ void copy_f32_kernel(const float* __restrict__ s,
                                float* __restrict__ d, int n) {
  int t = blockIdx.x * blockDim.x + threadIdx.x;
  if (t < n) d[t] = s[t];
}

__global__ void inds_to_float_kernel(const int* __restrict__ inds,
                                     float* __restrict__ out,
                                     int take, int stride, int total) {
  int t = blockIdx.x * blockDim.x + threadIdx.x;
  if (t >= total) return;
  int b = t / take, i = t % take;
  out[t] = (float)inds[(size_t)b * stride + i];
}

// ---------------------------------------------------------------------------
// Host driver (graph-capture safe: only stream launches, ws bump allocator)
// ---------------------------------------------------------------------------
extern "C" void kernel_launch(void* const* d_in, const int* in_sizes, int n_in,
                              void* d_out, int out_size, void* d_ws, size_t ws_size,
                              hipStream_t stream) {
  (void)in_sizes; (void)n_in; (void)out_size; (void)ws_size;
  const int B = 2;
  const float* pc = (const float*)d_in[0];

  // Parameter pointers (depth-first flatten of the params dict)
  const float* saW[4][2][3]; const float* saB[4][2][3];
  int ii = 1;
  for (int l = 0; l < 4; ++l)
    for (int s = 0; s < 2; ++s)
      for (int k = 0; k < 3; ++k) {
        saW[l][s][k] = (const float*)d_in[ii++];
        saB[l][s][k] = (const float*)d_in[ii++];
      }
  const float* fpW[2][2]; const float* fpB[2][2];
  for (int m = 0; m < 2; ++m)
    for (int k = 0; k < 2; ++k) {
      fpW[m][k] = (const float*)d_in[ii++];
      fpB[m][k] = (const float*)d_in[ii++];
    }

  static const int   NP[4]       = {4096, 1024, 512, 256};
  static const float RAD[4][2]   = {{0.1f,0.5f},{0.5f,1.0f},{1.0f,2.0f},{2.0f,4.0f}};
  static const int   NS[4][2]    = {{64,64},{32,32},{16,16},{16,16}};
  static const int   MLP[4][2][4]= {{{3,16,16,32},{3,32,32,64}},
                                    {{99,64,64,128},{99,64,96,128}},
                                    {{259,128,196,256},{259,128,196,256}},
                                    {{515,256,256,512},{515,256,384,512}}};
  static const int   FPD[2][3]   = {{1536,512,512},{768,512,512}};
  auto P64 = [](int x) { return (x + 63) & ~63; };

  // Workspace bump allocator (256B aligned)
  size_t off = 0;
  auto alloc = [&](size_t bytes) -> char* {
    off = (off + 255) & ~(size_t)255;
    char* p = (char*)d_ws + off;
    off += bytes;
    return p;
  };
  float* distbuf = (float*)alloc(sizeof(float) * (size_t)B * 20000);
  int*   inds[4]; float* nxyz[4]; float* feats[4]; int featC[4];
  for (int l = 0; l < 4; ++l) {
    inds[l]  = (int*)alloc(sizeof(int) * (size_t)B * NP[l]);
    nxyz[l]  = (float*)alloc(sizeof(float) * (size_t)B * NP[l] * 3);
    featC[l] = MLP[l][0][3] + MLP[l][1][3];
    feats[l] = (float*)alloc(sizeof(float) * (size_t)B * NP[l] * featC[l]);
  }
  int* idxbuf = (int*)alloc(sizeof(int) * 1024 * 64);
  const size_t BUFN = 1024ull * 64 * 64;            // largest padded intermediate
  float* bufA = (float*)alloc(sizeof(float) * BUFN);
  float* bufB = (float*)alloc(sizeof(float) * BUFN);
  float* ffp1 = (float*)alloc(sizeof(float) * (size_t)B * 512 * 512);

  // ---- Pre-swizzle all weights to padded bf16 B-fragment layout ----
  __bf16* wswSA[4][2][3];
  for (int l = 0; l < 4; ++l)
    for (int s = 0; s < 2; ++s)
      for (int ly = 0; ly < 3; ++ly) {
        int K = MLP[l][s][ly], N = MLP[l][s][ly + 1];
        int Kp = P64(K), Npp = P64(N);
        size_t elems = (size_t)Kp * Npp;
        wswSA[l][s][ly] = (__bf16*)alloc(elems * sizeof(__bf16));
        pack_w_kernel<<<dim3((unsigned)((elems + 255) / 256)), dim3(256), 0, stream>>>(
            saW[l][s][ly], wswSA[l][s][ly], K, N, Kp, Npp, (int)elems);
      }
  __bf16* wswFP[2][2];
  for (int m = 0; m < 2; ++m)
    for (int ly = 0; ly < 2; ++ly) {
      int K = FPD[m][ly], N = FPD[m][ly + 1];
      int Kp = P64(K), Npp = P64(N);
      size_t elems = (size_t)Kp * Npp;
      wswFP[m][ly] = (__bf16*)alloc(elems * sizeof(__bf16));
      pack_w_kernel<<<dim3((unsigned)((elems + 255) / 256)), dim3(256), 0, stream>>>(
          fpW[m][ly], wswFP[m][ly], K, N, Kp, Npp, (int)elems);
    }

  // ---- Set abstraction levels ----
  const float* curX = pc;
  const float* curF = nullptr;
  int curN = 20000, curC = 0;

  for (int l = 0; l < 4; ++l) {
    int npoint = NP[l];
    fps_kernel<<<dim3(B), dim3(256), 0, stream>>>(curX, curN, npoint, inds[l], distbuf);
    int tot = B * npoint;
    gather_xyz_kernel<<<dim3((tot + 255) / 256), dim3(256), 0, stream>>>(
        curX, inds[l], nxyz[l], npoint, curN, tot);

    int scaleOff = 0;
    for (int s = 0; s < 2; ++s) {
      float r = RAD[l][s];
      int ns = NS[l][s];
      const int* D = MLP[l][s];
      int CH = npoint > 1024 ? 1024 : npoint;       // chunk query points
      for (int b = 0; b < B; ++b) {
        const float* bx = curX + (size_t)b * curN * 3;
        const float* bf = curF ? curF + (size_t)b * curN * curC : nullptr;
        for (int p0 = 0; p0 < npoint; p0 += CH) {
          int P = CH;
          const float* qx = nxyz[l] + ((size_t)b * npoint + p0) * 3;
          ball_query_kernel<<<dim3((P + 127) / 128), dim3(128), 0, stream>>>(
              bx, qx, curN, P, r * r, ns, idxbuf);
          int tg = P * ns;
          group_kernel<<<dim3((tg + 255) / 256), dim3(256), 0, stream>>>(
              bx, bf, idxbuf, qx, bufA, P, ns, curC, P64(D[0]));
          // shared MLP: 3 layers of GEMM+bias+ReLU on (P*ns, Kp) padded tiles
          float* hin = bufA; float* hout = bufB;
          int M = P * ns;
          for (int ly = 0; ly < 3; ++ly) {
            int Kp  = P64(D[ly]);
            int N   = D[ly + 1];
            int Npp = P64(N);
            gemm_bias_relu_wmma<<<dim3(M / 64, Npp / 64), dim3(128), 0, stream>>>(
                hin, wswSA[l][s][ly], saB[l][s][ly], hout, Kp, N, Npp);
            float* t = hin; hin = hout; hout = t;
          }
          int Co = D[3];
          float* outp = feats[l] + ((size_t)b * npoint + p0) * featC[l] + scaleOff;
          int tp = P * Co;
          maxpool_kernel<<<dim3((tp + 255) / 256), dim3(256), 0, stream>>>(
              hin, outp, P, ns, Co, P64(Co), featC[l]);
        }
      }
      scaleOff += D[3];
    }
    curX = nxyz[l]; curF = feats[l]; curN = npoint; curC = featC[l];
  }

  // ---- Feature propagation ----
  float* fout = (float*)d_out;                        // (B,1024,512)
  for (int b = 0; b < B; ++b) {
    { // FP1: level3 (512 pts, 512ch) <- level4 (256 pts, 1024ch); MLP 1536->512->512
      int U = 512, Kn = 256, Cu = 512, Ck = 1024;
      fp_interp_concat_kernel<<<dim3((U + 63) / 64), dim3(64), 0, stream>>>(
          nxyz[2] + (size_t)b * U * 3, nxyz[3] + (size_t)b * Kn * 3,
          feats[2] + (size_t)b * U * Cu, feats[3] + (size_t)b * Kn * Ck,
          bufA, U, Kn, Cu, Ck);
      gemm_bias_relu_wmma<<<dim3(U / 64, 512 / 64), dim3(128), 0, stream>>>(
          bufA, wswFP[0][0], fpB[0][0], bufB, 1536, 512, 512);
      gemm_bias_relu_wmma<<<dim3(U / 64, 512 / 64), dim3(128), 0, stream>>>(
          bufB, wswFP[0][1], fpB[0][1], ffp1 + (size_t)b * 512 * 512, 512, 512, 512);
    }
    { // FP2: level2 (1024 pts, 256ch) <- level3 (512 pts, FP1 512ch); MLP 768->512->512
      int U = 1024, Kn = 512, Cu = 256, Ck = 512;
      fp_interp_concat_kernel<<<dim3((U + 63) / 64), dim3(64), 0, stream>>>(
          nxyz[1] + (size_t)b * U * 3, nxyz[2] + (size_t)b * Kn * 3,
          feats[1] + (size_t)b * U * Cu, ffp1 + (size_t)b * Kn * Ck,
          bufA, U, Kn, Cu, Ck);
      gemm_bias_relu_wmma<<<dim3(U / 64, 512 / 64), dim3(128), 0, stream>>>(
          bufA, wswFP[1][0], fpB[1][0], bufB, 768, 512, 512);
      gemm_bias_relu_wmma<<<dim3(U / 64, 512 / 64), dim3(128), 0, stream>>>(
          bufB, wswFP[1][1], fpB[1][1], fout + (size_t)b * 1024 * 512, 512, 512, 512);
    }
  }

  // ---- Remaining tuple outputs: sa_xyz[1] and fp2_inds ----
  float* oxyz = fout + (size_t)B * 1024 * 512;
  copy_f32_kernel<<<dim3((B * 1024 * 3 + 255) / 256), dim3(256), 0, stream>>>(
      nxyz[1], oxyz, B * 1024 * 3);
  float* oind = oxyz + (size_t)B * 1024 * 3;
  inds_to_float_kernel<<<dim3((B * 1024 + 255) / 256), dim3(256), 0, stream>>>(
      inds[0], oind, 1024, 4096, B * 1024);
}
